// Pair_construction_layer_25451976196778
// MI455X (gfx1250) — compile-verified, matrix operand order checked by reading, not run on hardware
//
#include <hip/hip_runtime.h>
#include <hip/hip_bf16.h>
#include <math.h>

// ---------------------------------------------------------------------------
// Pair-construction equivariant layer for MI455X (gfx1250, wave32, WMMA).
// All dense math runs through v_wmma_f32_16x16x32_bf16 (fp32 accumulate).
// GEMM staging: double-buffered GLOBAL_LOAD_ASYNC_TO_LDS (ASYNCcnt) when the
// toolchain exposes the builtin, else sync global->VGPR->LDS staging.
// ---------------------------------------------------------------------------

typedef __bf16 bf16_t;
typedef __bf16 v16bf __attribute__((ext_vector_type(16)));
typedef __bf16 bf16x8 __attribute__((ext_vector_type(8)));
typedef float  v8f   __attribute__((ext_vector_type(8)));
typedef int    v4i   __attribute__((ext_vector_type(4)));

#define NN   32768
#define MUL  256
#define DD   1024

// GEMM tiling: 128x128 block tile, BK=64, 256 threads = 8 waves, each wave 32x64.
#define TM   128
#define TN   128
#define TK   64
#define LDSK (TK + 8)   // bf16 row stride in LDS: 144B = 9*16B -> conflict-free b128

#if defined(__gfx1250__) && __has_builtin(__builtin_amdgcn_global_load_async_to_lds_b128)
#define HAVE_ASYNC_LDS 1
#else
#define HAVE_ASYNC_LDS 0
#endif

typedef __attribute__((address_space(1))) v4i as1_v4i;   // global 16B chunk
typedef __attribute__((address_space(3))) v4i as3_v4i;   // LDS 16B chunk

static __device__ __forceinline__ void wait_async0()
{
#if defined(__gfx1250__)
#if __has_builtin(__builtin_amdgcn_s_wait_asynccnt)
    __builtin_amdgcn_s_wait_asynccnt(0);
#else
    asm volatile("s_wait_asynccnt 0x0" ::: "memory");
#endif
#endif
}

// Epilogue modes
#define EP_SILU_BF16   0   // out_bf16 = silu(acc + bias)        (MLP layer 1)
#define EP_BIAS_BF16   1   // out_bf16 = acc (+ bias if nonnull) (MLP layer 2, linears)
#define EP_FINAL_SCAL  2   // d_out[row*1024+col]          = acc + bias + old_fii
#define EP_FINAL_VEC   3   // d_out[row*1024+256+3col+cmp] = acc + old_fii

template<int KDIM, int MODE>
__global__ __launch_bounds__(256, 2)
void gemm_bf16_wmma(const bf16_t* __restrict__ A,    // [rows, KDIM] bf16
                    const bf16_t* __restrict__ Wt,   // [KDIM, KDIM] bf16, N-major (row=out col)
                    const float*  __restrict__ bias, // [KDIM] or null
                    void*         __restrict__ outp,
                    const float*  __restrict__ oldf, // only FINAL modes
                    int comp)
{
#if HAVE_ASYNC_LDS
    __shared__ bf16_t lA[2][TM * LDSK];
    __shared__ bf16_t lB[2][TN * LDSK];
#else
    __shared__ bf16_t lA[1][TM * LDSK];
    __shared__ bf16_t lB[1][TN * LDSK];
#endif

    const int tid  = threadIdx.x;
    const int lane = tid & 31;
    const int wave = tid >> 5;
    const int bm   = blockIdx.x * TM;
    const int bn   = blockIdx.y * TN;
    const int wm   = (wave >> 1) * 32;   // 4 waves along M
    const int wn   = (wave & 1) * 64;    // 2 waves along N
    const int fr   = lane & 15;          // fragment row (A) / col (B)
    const int kq   = (lane >> 4) * 8;    // K sub-offset per 16-bit WMMA layout

    v8f acc[2][4];
    #pragma unroll
    for (int i = 0; i < 2; ++i)
        #pragma unroll
        for (int j = 0; j < 4; ++j)
            #pragma unroll
            for (int e = 0; e < 8; ++e) acc[i][j][e] = 0.0f;

    // 2 K-steps of v_wmma_f32_16x16x32_bf16 over one staged 128x64 tile pair
    auto compute_tile = [&](const bf16_t* __restrict__ tA,
                            const bf16_t* __restrict__ tB) {
        #pragma unroll
        for (int k0 = 0; k0 < TK; k0 += 32) {
            v16bf af[2], bfr[4];
            #pragma unroll
            for (int i = 0; i < 2; ++i) {
                const bf16_t* p = tA + (size_t)(wm + i * 16 + fr) * LDSK + k0 + kq;
                union { v16bf v; bf16x8 h[2]; } u;
                u.h[0] = *(const bf16x8*)p;
                u.h[1] = *(const bf16x8*)(p + 16);
                af[i] = u.v;
            }
            #pragma unroll
            for (int j = 0; j < 4; ++j) {
                const bf16_t* p = tB + (size_t)(wn + j * 16 + fr) * LDSK + k0 + kq;
                union { v16bf v; bf16x8 h[2]; } u;
                u.h[0] = *(const bf16x8*)p;
                u.h[1] = *(const bf16x8*)(p + 16);
                bfr[j] = u.v;
            }
            #pragma unroll
            for (int i = 0; i < 2; ++i)
                #pragma unroll
                for (int j = 0; j < 4; ++j)
                    acc[i][j] = __builtin_amdgcn_wmma_f32_16x16x32_bf16(
                        false, af[i], false, bfr[j], (short)0, acc[i][j], false, false);
        }
    };

#if HAVE_ASYNC_LDS
    // ---- double-buffered async staging: ASYNCcnt-tracked LDS DMA ----------
    auto stage_async = [&](int kt, int buf) {
        #pragma unroll
        for (int c = 0; c < 4; ++c) {
            int chunk = tid + c * 256;          // 1024 chunks of 16B per tile
            int row   = chunk >> 3;
            int col8  = (chunk & 7) * 8;
            const bf16_t* ga = A  + (size_t)(bm + row) * KDIM + kt + col8;
            const bf16_t* gb = Wt + (size_t)(bn + row) * KDIM + kt + col8;
            __builtin_amdgcn_global_load_async_to_lds_b128(
                (as1_v4i*)ga, (as3_v4i*)&lA[buf][row * LDSK + col8], 0, 0);
            __builtin_amdgcn_global_load_async_to_lds_b128(
                (as1_v4i*)gb, (as3_v4i*)&lB[buf][row * LDSK + col8], 0, 0);
        }
    };

    stage_async(0, 0);
    for (int kt = 0; kt < KDIM; kt += TK) {
        const int buf = (kt / TK) & 1;
        wait_async0();                 // own async loads for `buf` complete
        __syncthreads();               // -> everyone's loads complete / prev reads done
        if (kt + TK < KDIM) stage_async(kt + TK, buf ^ 1);
        compute_tile(lA[buf], lB[buf]);
    }
#else
    // ---- fallback: sync staging through VGPRs + prefetch ------------------
    for (int kt = 0; kt < KDIM; kt += TK) {
        #pragma unroll
        for (int c = 0; c < 4; ++c) {
            int chunk = tid + c * 256;
            int row   = chunk >> 3;
            int col8  = (chunk & 7) * 8;
            const bf16_t* ga = A  + (size_t)(bm + row) * KDIM + kt + col8;
            const bf16_t* gb = Wt + (size_t)(bn + row) * KDIM + kt + col8;
            *(bf16x8*)(&lA[0][row * LDSK + col8]) = *(const bf16x8*)ga;
            *(bf16x8*)(&lB[0][row * LDSK + col8]) = *(const bf16x8*)gb;
            if (kt + TK < KDIM) {       // global_prefetch_b8 next K-tile
                __builtin_prefetch(ga + TK, 0, 1);
                __builtin_prefetch(gb + TK, 0, 1);
            }
        }
        __syncthreads();
        compute_tile(lA[0], lB[0]);
        __syncthreads();
    }
#endif

    // -------- epilogue -----------------------------------------------------
    #pragma unroll
    for (int i = 0; i < 2; ++i) {
        #pragma unroll
        for (int j = 0; j < 4; ++j) {
            #pragma unroll
            for (int r = 0; r < 8; ++r) {
                int row = bm + wm + i * 16 + r + ((lane >> 4) << 3);
                int col = bn + wn + j * 16 + (lane & 15);
                float v = acc[i][j][r];
                if (MODE == EP_SILU_BF16) {
                    v += bias[col];
                    v  = v / (1.0f + __expf(-v));
                    ((bf16_t*)outp)[(size_t)row * KDIM + col] = (bf16_t)v;
                } else if (MODE == EP_BIAS_BF16) {
                    if (bias) v += bias[col];
                    ((bf16_t*)outp)[(size_t)row * KDIM + col] = (bf16_t)v;
                } else if (MODE == EP_FINAL_SCAL) {
                    size_t o = (size_t)row * DD + col;
                    ((float*)outp)[o] = v + bias[col] + oldf[o];
                } else { // EP_FINAL_VEC
                    size_t o = (size_t)row * DD + MUL + 3 * (size_t)col + comp;
                    ((float*)outp)[o] = v + oldf[o];
                }
            }
        }
    }
}

// ---------------------------------------------------------------------------
// Elementwise kernels
// ---------------------------------------------------------------------------

// convert fp32 weight [R,C] -> bf16 transposed [C,R] (N-major), optional scale
__global__ void wprep_kernel(const float* __restrict__ src, bf16_t* __restrict__ dst,
                             int R, int C, float scale)
{
    int idx = blockIdx.x * 256 + threadIdx.x;
    if (idx >= R * C) return;
    int j = idx / R;          // dst row  = output column
    int k = idx - j * R;      // dst col  = input dim
    dst[idx] = (bf16_t)(src[(size_t)k * C + j] * scale);
}

// f0 = [scalars, |vectors|] from interleaved x [N,1024] -> bf16 [N,512]
__global__ void f0_kernel(const float* __restrict__ x, bf16_t* __restrict__ f0)
{
    int idx = blockIdx.x * 256 + threadIdx.x;   // N*MUL threads
    int n = idx >> 8, u = idx & 255;
    const float* xr = x + (size_t)n * DD;
    float vx = xr[MUL + 3 * u], vy = xr[MUL + 3 * u + 1], vz = xr[MUL + 3 * u + 2];
    bf16_t* fr = f0 + (size_t)n * 512;
    fr[u]       = (bf16_t)xr[u];
    fr[256 + u] = (bf16_t)sqrtf(vx * vx + vy * vy + vz * vz + 1e-12f);
}

// gate: As = g_s ; Av_i = x_v[:,:,i] * g_v   (bf16 planes [N,256])
__global__ void gate_kernel(const float* __restrict__ x, const bf16_t* __restrict__ g,
                            bf16_t* __restrict__ As, bf16_t* __restrict__ Avx,
                            bf16_t* __restrict__ Avy, bf16_t* __restrict__ Avz)
{
    int idx = blockIdx.x * 256 + threadIdx.x;
    int n = idx >> 8, u = idx & 255;
    const bf16_t* gr = g + (size_t)n * 512;
    float gv = (float)gr[256 + u];
    const float* xv = x + (size_t)n * DD + MUL + 3 * u;
    As [idx] = gr[u];
    Avx[idx] = (bf16_t)(xv[0] * gv);
    Avy[idx] = (bf16_t)(xv[1] * gv);
    Avz[idx] = (bf16_t)(xv[2] * gv);
}

// 'uuu' CG tensor product + residual x, outputs bf16 planes ts, tv{x,y,z}
__global__ void tp_kernel(const bf16_t* __restrict__ s1,  const bf16_t* __restrict__ v1x,
                          const bf16_t* __restrict__ v1y, const bf16_t* __restrict__ v1z,
                          const bf16_t* __restrict__ s2,  const bf16_t* __restrict__ v2x,
                          const bf16_t* __restrict__ v2y, const bf16_t* __restrict__ v2z,
                          const float* __restrict__ x, const float* __restrict__ tpw,
                          bf16_t* __restrict__ ts, bf16_t* __restrict__ tvx,
                          bf16_t* __restrict__ tvy, bf16_t* __restrict__ tvz)
{
    const float INV3 = 0.57735026918962576f;   // 1/sqrt(3)
    const float INV6 = 0.40824829046386302f;   // 1/sqrt(6)
    const float A0   = 0.44721359549995794f;   // sqrt(1/5)
    const float A1   = 0.77459666924148338f;   // sqrt(3/5)
    int idx = blockIdx.x * 256 + threadIdx.x;
    int n = idx >> 8, u = idx & 255;
    float w0 = tpw[u], w1 = tpw[256 + u], w2 = tpw[512 + u], w3 = tpw[768 + u], w4 = tpw[1024 + u];
    float S1 = (float)s1[idx], S2 = (float)s2[idx];
    float ax = (float)v1x[idx], ay = (float)v1y[idx], az = (float)v1z[idx];
    float bx = (float)v2x[idx], by = (float)v2y[idx], bz = (float)v2z[idx];
    float dot = ax * bx + ay * by + az * bz;
    float cx = ay * bz - az * by, cy = az * bx - ax * bz, cz = ax * by - ay * bx;
    const float* xr = x + (size_t)n * DD;
    float o0 = A0 * (w0 * S1 * S2 + w3 * INV3 * dot) + xr[u];
    float c1 = A1 * INV3 * w1 * S1;   // * v2
    float c2 = A1 * INV3 * w2 * S2;   // * v1
    float c4 = A1 * INV6 * w4;        // * cross
    ts [idx] = (bf16_t)o0;
    tvx[idx] = (bf16_t)(c1 * bx + c2 * ax + c4 * cx + xr[MUL + 3 * u]);
    tvy[idx] = (bf16_t)(c1 * by + c2 * ay + c4 * cy + xr[MUL + 3 * u + 1]);
    tvz[idx] = (bf16_t)(c1 * bz + c2 * az + c4 * cz + xr[MUL + 3 * u + 2]);
}

// f0 from bf16 planes (for the output NormGate)
__global__ void f0p_kernel(const bf16_t* __restrict__ ts,  const bf16_t* __restrict__ tvx,
                           const bf16_t* __restrict__ tvy, const bf16_t* __restrict__ tvz,
                           bf16_t* __restrict__ f0)
{
    int idx = blockIdx.x * 256 + threadIdx.x;
    int n = idx >> 8, u = idx & 255;
    float vx = (float)tvx[idx], vy = (float)tvy[idx], vz = (float)tvz[idx];
    bf16_t* fr = f0 + (size_t)n * 512;
    fr[u]       = ts[idx];
    fr[256 + u] = (bf16_t)sqrtf(vx * vx + vy * vy + vz * vz + 1e-12f);
}

// output gate: Aso = g_s ; Avo_i = tv_i * g_v
__global__ void gateo_kernel(const bf16_t* __restrict__ g,
                             const bf16_t* __restrict__ tvx, const bf16_t* __restrict__ tvy,
                             const bf16_t* __restrict__ tvz,
                             bf16_t* __restrict__ Aso, bf16_t* __restrict__ Avx,
                             bf16_t* __restrict__ Avy, bf16_t* __restrict__ Avz)
{
    int idx = blockIdx.x * 256 + threadIdx.x;
    int n = idx >> 8, u = idx & 255;
    const bf16_t* gr = g + (size_t)n * 512;
    float gv = (float)gr[256 + u];
    Aso[idx] = gr[u];
    Avx[idx] = (bf16_t)((float)tvx[idx] * gv);
    Avy[idx] = (bf16_t)((float)tvy[idx] * gv);
    Avz[idx] = (bf16_t)((float)tvz[idx] * gv);
}

// ---------------------------------------------------------------------------
// Launcher
// ---------------------------------------------------------------------------
extern "C" void kernel_launch(void* const* d_in, const int* in_sizes, int n_in,
                              void* d_out, int out_size, void* d_ws, size_t ws_size,
                              hipStream_t stream)
{
    const float* x    = (const float*)d_in[0];
    const float* oldf = (const float*)d_in[1];
    const float* ngw1[3] = { (const float*)d_in[2],  (const float*)d_in[6],  (const float*)d_in[10] };
    const float* ngb1[3] = { (const float*)d_in[3],  (const float*)d_in[7],  (const float*)d_in[11] };
    const float* ngw2[3] = { (const float*)d_in[4],  (const float*)d_in[8],  (const float*)d_in[12] };
    const float* ngb2[3] = { (const float*)d_in[5],  (const float*)d_in[9],  (const float*)d_in[13] };
    const float* lw0[3]  = { (const float*)d_in[14], (const float*)d_in[17], (const float*)d_in[20] };
    const float* lb0[3]  = { (const float*)d_in[15], (const float*)d_in[18], (const float*)d_in[21] };
    const float* lw1[3]  = { (const float*)d_in[16], (const float*)d_in[19], (const float*)d_in[22] };
    const float* tpw     = (const float*)d_in[23];

    // ---- workspace layout: 4MB bf16 weights + 16 planes of 16MB -----------
    bf16_t* WT = (bf16_t*)d_ws;
    bf16_t* ngw1t[3], * ngw2t[3], * lw0t[3], * lw1t[3];
    size_t off = 0;
    for (int p = 0; p < 3; ++p) { ngw1t[p] = WT + off; off += 512 * 512;
                                  ngw2t[p] = WT + off; off += 512 * 512; }
    for (int p = 0; p < 3; ++p) { lw0t[p]  = WT + off; off += 256 * 256;
                                  lw1t[p]  = WT + off; off += 256 * 256; }
    auto U = [&](int p) -> bf16_t* {
        return (bf16_t*)((char*)d_ws + (4ull << 20) + (size_t)p * (16ull << 20));
    };

    const int   NU   = NN * MUL;           // elementwise thread count
    const int   EB   = NU / 256;           // elementwise blocks
    const float ILIN = 1.0f / 16.0f;       // 1/sqrt(MUL)
    dim3 g512(NN / TM, 512 / TN), g256(NN / TM, 256 / TN), blk(256);

    // 1) weight conversion + transpose to bf16 N-major
    for (int p = 0; p < 3; ++p) {
        wprep_kernel<<<1024, blk, 0, stream>>>(ngw1[p], ngw1t[p], 512, 512, 1.0f);
        wprep_kernel<<<1024, blk, 0, stream>>>(ngw2[p], ngw2t[p], 512, 512, 1.0f);
        wprep_kernel<<<256,  blk, 0, stream>>>(lw0[p],  lw0t[p],  256, 256, ILIN);
        wprep_kernel<<<256,  blk, 0, stream>>>(lw1[p],  lw1t[p],  256, 256, ILIN);
    }

    // 2) shared NormGate input features f0(x) -> U0-1
    f0_kernel<<<EB, blk, 0, stream>>>(x, U(0));

    // 3) NormGate MLPs (side 1 & 2): h = silu(f0 W1 + b1); g = h W2 + b2
    gemm_bf16_wmma<512, EP_SILU_BF16><<<g512, blk, 0, stream>>>(U(0), ngw1t[0], ngb1[0], U(2), nullptr, 0);
    gemm_bf16_wmma<512, EP_SILU_BF16><<<g512, blk, 0, stream>>>(U(0), ngw1t[1], ngb1[1], U(4), nullptr, 0);
    gemm_bf16_wmma<512, EP_BIAS_BF16><<<g512, blk, 0, stream>>>(U(2), ngw2t[0], ngb2[0], U(0), nullptr, 0);
    gemm_bf16_wmma<512, EP_BIAS_BF16><<<g512, blk, 0, stream>>>(U(4), ngw2t[1], ngb2[1], U(6), nullptr, 0);

    // 4) gating -> contiguous bf16 A-planes for the irrep linears
    gate_kernel<<<EB, blk, 0, stream>>>(x, U(0), U(2), U(3), U(4), U(5));    // side 1
    gate_kernel<<<EB, blk, 0, stream>>>(x, U(6), U(8), U(9), U(10), U(11));  // side 2

    // 5) irrep linears 1 & 2 (scalar + 3 vector-component GEMMs each)
    gemm_bf16_wmma<256, EP_BIAS_BF16><<<g256, blk, 0, stream>>>(U(2),  lw0t[0], lb0[0], U(0),  nullptr, 0);
    gemm_bf16_wmma<256, EP_BIAS_BF16><<<g256, blk, 0, stream>>>(U(3),  lw1t[0], nullptr, U(1),  nullptr, 0);
    gemm_bf16_wmma<256, EP_BIAS_BF16><<<g256, blk, 0, stream>>>(U(4),  lw1t[0], nullptr, U(6),  nullptr, 0);
    gemm_bf16_wmma<256, EP_BIAS_BF16><<<g256, blk, 0, stream>>>(U(5),  lw1t[0], nullptr, U(7),  nullptr, 0);
    gemm_bf16_wmma<256, EP_BIAS_BF16><<<g256, blk, 0, stream>>>(U(8),  lw0t[1], lb0[1], U(12), nullptr, 0);
    gemm_bf16_wmma<256, EP_BIAS_BF16><<<g256, blk, 0, stream>>>(U(9),  lw1t[1], nullptr, U(13), nullptr, 0);
    gemm_bf16_wmma<256, EP_BIAS_BF16><<<g256, blk, 0, stream>>>(U(10), lw1t[1], nullptr, U(14), nullptr, 0);
    gemm_bf16_wmma<256, EP_BIAS_BF16><<<g256, blk, 0, stream>>>(U(11), lw1t[1], nullptr, U(15), nullptr, 0);

    // 6) CG tensor product + residual -> t planes (U2..U5)
    tp_kernel<<<EB, blk, 0, stream>>>(U(0), U(1), U(6), U(7),
                                      U(12), U(13), U(14), U(15),
                                      x, tpw, U(2), U(3), U(4), U(5));

    // 7) output NormGate on t
    f0p_kernel<<<EB, blk, 0, stream>>>(U(2), U(3), U(4), U(5), U(8));
    gemm_bf16_wmma<512, EP_SILU_BF16><<<g512, blk, 0, stream>>>(U(8),  ngw1t[2], ngb1[2], U(10), nullptr, 0);
    gemm_bf16_wmma<512, EP_BIAS_BF16><<<g512, blk, 0, stream>>>(U(10), ngw2t[2], ngb2[2], U(12), nullptr, 0);
    gateo_kernel<<<EB, blk, 0, stream>>>(U(12), U(3), U(4), U(5), U(14), U(15), U(0), U(1));

    // 8) final irrep linear, fused with old_fii residual, scattered into d_out
    gemm_bf16_wmma<256, EP_FINAL_SCAL><<<g256, blk, 0, stream>>>(U(14), lw0t[2], lb0[2], d_out, oldf, 0);
    gemm_bf16_wmma<256, EP_FINAL_VEC ><<<g256, blk, 0, stream>>>(U(15), lw1t[2], nullptr, d_out, oldf, 0);
    gemm_bf16_wmma<256, EP_FINAL_VEC ><<<g256, blk, 0, stream>>>(U(0),  lw1t[2], nullptr, d_out, oldf, 1);
    gemm_bf16_wmma<256, EP_FINAL_VEC ><<<g256, blk, 0, stream>>>(U(1),  lw1t[2], nullptr, d_out, oldf, 2);

    (void)in_sizes; (void)n_in; (void)out_size; (void)ws_size;
}
